// Framework_82248623718499
// MI455X (gfx1250) — matrix operand order repaired and checked by reading, not compiled, not run on hardware
//
#include <hip/hip_runtime.h>
#include <hip/hip_bf16.h>
#include <math.h>
#include <stdint.h>

typedef __attribute__((ext_vector_type(16))) _Float16 v16h;
typedef __attribute__((ext_vector_type(8)))  float    v8f;
typedef __attribute__((ext_vector_type(2)))  float    v2f;
typedef int v4i_async __attribute__((vector_size(16)));

union AFragH { v16h v; _Float16 e[16]; uint4 q[2]; };
union CFragF { v8f  v; float    e[8];  };

#define D_DIM 128
#define H_DIM 64

// ---- async global->LDS copy (CDNA5): builtin if present, else inline asm ----
#if __has_builtin(__builtin_amdgcn_global_load_async_to_lds_b128)
#define ASYNC_COPY_B128(SRC, DST)                                              \
  __builtin_amdgcn_global_load_async_to_lds_b128(                              \
      (__attribute__((address_space(1))) v4i_async*)(uintptr_t)(SRC),          \
      (__attribute__((address_space(3))) v4i_async*)(uint32_t)(uintptr_t)(DST),\
      0, 0)
#else
#define ASYNC_COPY_B128(SRC, DST)                                              \
  asm volatile("global_load_async_to_lds_b128 %0, %1, off"                     \
               :: "v"((uint32_t)(uintptr_t)(DST)), "v"((const void*)(SRC))     \
               : "memory")
#endif

#if __has_builtin(__builtin_amdgcn_s_wait_asynccnt)
#define WAIT_ASYNC(N) __builtin_amdgcn_s_wait_asynccnt(N)
#else
#define WAIT_ASYNC(N)                                                          \
  do {                                                                         \
    if ((N) == 0) asm volatile("s_wait_asynccnt 0x0" ::: "memory");            \
    else          asm volatile("s_wait_asynccnt 0x10" ::: "memory");           \
  } while (0)
#endif

// ------------------------------------------------------------------
// prep: W1 [D,H] f32  ->  W1t [H,D] f16 (transposed, fragment friendly)
// ------------------------------------------------------------------
__global__ void prep_w1t(const float* __restrict__ W1, _Float16* __restrict__ W1t) {
  int i = blockIdx.x * blockDim.x + threadIdx.x;
  if (i < D_DIM * H_DIM) {
    int k = i / H_DIM, n = i % H_DIM;
    W1t[n * D_DIM + k] = (_Float16)W1[i];
  }
}

// ------------------------------------------------------------------
// attention scores: per 16-row tile, S = tanh(X @ W1 + b1) @ W2 + b2
// - W1t fragments hoisted into 128 VGPRs (loop invariant)
// - x tiles gathered with async global->LDS DMA, double buffered
// ------------------------------------------------------------------
__launch_bounds__(128)
__global__ void attn_scores(const float* __restrict__ feat,
                            const int*  __restrict__ fidx,
                            const _Float16* __restrict__ W1t,
                            const float* __restrict__ b1,
                            const float* __restrict__ W2,
                            const float* __restrict__ b2,
                            float* __restrict__ scores,
                            int M)
{
  // 4 waves x double-buffered 16x128 f32 tiles = 64 KB
  __shared__ __align__(16) float sx[4][2][16][D_DIM];

  const int tid  = threadIdx.x;
  const int lane = tid & 31;
  const int wave = tid >> 5;
  const int half = lane >> 4;   // half-wave select
  const int sub  = lane & 15;   // row (A) / col (B,C) within tile

  // ---- hoist B fragments of W1t into registers: bw[ks][nt] ----
  // B (16-bit, 32x16): lane holds col = nt*16+sub; K chunks [kb+8h,+8) and [kb+8h+16,+8)
  v16h bw[4][4];
  #pragma unroll
  for (int ks = 0; ks < 4; ++ks) {
    const int kb = ks * 32;
    #pragma unroll
    for (int t = 0; t < 4; ++t) {
      const uint4* bp = (const uint4*)&W1t[(size_t)(t * 16 + sub) * D_DIM];
      AFragH tmp;
      tmp.q[0] = bp[(kb + 8 * half) >> 3];
      tmp.q[1] = bp[(kb + 8 * half + 16) >> 3];
      bw[ks][t] = tmp.v;
    }
  }
  // per-lane slice of b1 / W2 (columns t*16+sub)
  float rb1[4], rW2[4];
  #pragma unroll
  for (int t = 0; t < 4; ++t) {
    rb1[t] = b1[t * 16 + sub];
    rW2[t] = W2[t * 16 + sub];
  }
  const float bias2 = b2[0];

  const int numTiles = (M + 15) >> 4;
  const int stride   = gridDim.x * 4;
  int t = blockIdx.x * 4 + wave;

  // prologue: kick off gather of first tile into buffer 0
  if (t < numTiles) {
    const int rowBase = t * 16;
    #pragma unroll
    for (int r = 0; r < 16; ++r) {
      int rr = rowBase + r;
      int g  = fidx[(rr < M) ? rr : (M - 1)];
      const float* src = feat + (size_t)g * D_DIM + lane * 4;
      float*       dst = &sx[wave][0][r][lane * 4];
      ASYNC_COPY_B128(src, dst);
    }
  }

  int buf = 0;
  for (; t < numTiles; t += stride) {
    const int tn = t + stride;
    // issue next tile's gather into the other buffer
    if (tn < numTiles) {
      const int rowBase = tn * 16;
      #pragma unroll
      for (int r = 0; r < 16; ++r) {
        int rr = rowBase + r;
        int g  = fidx[(rr < M) ? rr : (M - 1)];
        const float* src = feat + (size_t)g * D_DIM + lane * 4;
        float*       dst = &sx[wave][buf ^ 1][r][lane * 4];
        ASYNC_COPY_B128(src, dst);
      }
      WAIT_ASYNC(16);   // 16 newest (next tile) may remain in flight
    } else {
      WAIT_ASYNC(0);
    }

    // ---- compute current tile from LDS (f32 -> f16 A fragments) ----
    CFragF acc[4];
    #pragma unroll
    for (int q = 0; q < 4; ++q)
      #pragma unroll
      for (int j = 0; j < 8; ++j) acc[q].e[j] = 0.f;

    const float* rp = &sx[wave][buf][sub][0];
    #pragma unroll
    for (int ks = 0; ks < 4; ++ks) {
      const int kb = ks * 32;
      // A (16-bit, 16x32): lane holds row=sub; K chunks [kb+8h,+8) and [kb+8h+16,+8)
      float4 c0 = *(const float4*)(rp + kb + 8 * half);
      float4 c1 = *(const float4*)(rp + kb + 8 * half + 4);
      float4 c2 = *(const float4*)(rp + kb + 8 * half + 16);
      float4 c3 = *(const float4*)(rp + kb + 8 * half + 20);
      AFragH a;
      a.e[0]  = (_Float16)c0.x; a.e[1]  = (_Float16)c0.y;
      a.e[2]  = (_Float16)c0.z; a.e[3]  = (_Float16)c0.w;
      a.e[4]  = (_Float16)c1.x; a.e[5]  = (_Float16)c1.y;
      a.e[6]  = (_Float16)c1.z; a.e[7]  = (_Float16)c1.w;
      a.e[8]  = (_Float16)c2.x; a.e[9]  = (_Float16)c2.y;
      a.e[10] = (_Float16)c2.z; a.e[11] = (_Float16)c2.w;
      a.e[12] = (_Float16)c3.x; a.e[13] = (_Float16)c3.y;
      a.e[14] = (_Float16)c3.z; a.e[15] = (_Float16)c3.w;
      #pragma unroll
      for (int q = 0; q < 4; ++q) {
        acc[q].v = __builtin_amdgcn_wmma_f32_16x16x32_f16(
            false, a.v, false, bw[ks][q], (short)0, acc[q].v, false, false);
      }
    }

    // epilogue: score[m] = sum_n tanh(S[m][n] + b1[n]) * W2[n] + b2
    const int rowBase = t * 16;
    #pragma unroll
    for (int r = 0; r < 8; ++r) {
      float v = 0.f;
      #pragma unroll
      for (int q = 0; q < 4; ++q)
        v += tanhf(acc[q].e[r] + rb1[q]) * rW2[q];
      #pragma unroll
      for (int off = 1; off < 16; off <<= 1)   // reduce across 16 cols
        v += __shfl_xor(v, off, 32);
      if (sub == 0) {
        int row = rowBase + half * 8 + r;      // lane0 -> rows 0..7, lane16 -> 8..15
        if (row < M) scores[row] = v + bias2;
      }
    }
    buf ^= 1;
  }
}

// ------------------------------------------------------------------
// per-molecule segment softmax + weighted pooling (batch_idx sorted)
// ------------------------------------------------------------------
__device__ __forceinline__ int lower_bound_i(const int* a, int n, int key) {
  int lo = 0, hi = n;
  while (lo < hi) {
    int mid = (lo + hi) >> 1;
    if (a[mid] < key) lo = mid + 1; else hi = mid;
  }
  return lo;
}

__launch_bounds__(128)
__global__ void seg_softmax_pool(const float* __restrict__ feat,
                                 const int*  __restrict__ fidx,
                                 const int*  __restrict__ bidx,
                                 const float* __restrict__ scores,
                                 float* __restrict__ pooled,
                                 int M)
{
  const int b = blockIdx.x;
  const int tid = threadIdx.x;           // == feature index d (0..127)
  __shared__ float red[128];
  __shared__ int seg[2];
  if (tid == 0) {
    seg[0] = lower_bound_i(bidx, M, b);
    seg[1] = lower_bound_i(bidx, M, b + 1);
  }
  __syncthreads();
  const int lo = seg[0], hi = seg[1];

  // segment max
  float pm = -3.402823466e38f;
  for (int j = lo + tid; j < hi; j += 128) pm = fmaxf(pm, scores[j]);
  red[tid] = pm; __syncthreads();
  for (int s = 64; s > 0; s >>= 1) {
    if (tid < s) red[tid] = fmaxf(red[tid], red[tid + s]);
    __syncthreads();
  }
  const float smax = red[0];
  __syncthreads();

  // segment sum of exp
  float ps = 0.f;
  for (int j = lo + tid; j < hi; j += 128) ps += expf(scores[j] - smax);
  red[tid] = ps; __syncthreads();
  for (int s = 64; s > 0; s >>= 1) {
    if (tid < s) red[tid] += red[tid + s];
    __syncthreads();
  }
  const float inv = 1.0f / (red[0] + 1e-16f);

  // pooled[b][d] = sum_j alpha_j * x[j][d]   (coalesced 512B row reads)
  float acc = 0.f;
  for (int j = lo; j < hi; ++j) {
    float w = expf(scores[j] - smax) * inv;
    acc += w * feat[(size_t)fidx[j] * D_DIM + tid];
  }
  pooled[(size_t)b * D_DIM + tid] = acc;
}

// ------------------------------------------------------------------
// predictor: logits = relu(pooled @ W3 + b3) @ W4 + b4
// f32 WMMA 16x16x4 (full precision)
// ------------------------------------------------------------------
__launch_bounds__(128)
__global__ void predictor(const float* __restrict__ pooled,
                          const float* __restrict__ W3,
                          const float* __restrict__ b3,
                          const float* __restrict__ W4,
                          const float* __restrict__ b4,
                          float* __restrict__ out,
                          int B, int T)
{
  __shared__ __align__(16) float tile[4][16][D_DIM + 4];  // padded: 33.8 KB
  const int tid  = threadIdx.x;
  const int lane = tid & 31;
  const int wave = tid >> 5;
  const int half = lane >> 4;
  const int sub  = lane & 15;
  const int numTiles = (B + 15) >> 4;

  for (int base = blockIdx.x * 4; base < numTiles; base += gridDim.x * 4) {
    const int t = base + wave;
    const bool active = (t < numTiles);
    const int rowBase = t * 16;

    if (active) {
      for (int r = 0; r < 16; ++r) {
        int rr = rowBase + r; if (rr >= B) rr = B - 1;
        float4 v = ((const float4*)(pooled + (size_t)rr * D_DIM))[lane];
        float* dp = &tile[wave][r][lane * 4];
        dp[0] = v.x; dp[1] = v.y; dp[2] = v.z; dp[3] = v.w;
      }
    }
    __syncthreads();

    if (active) {
      asm volatile("s_wait_dscnt 0x0" ::: "memory");

      // GEMM1: h = relu(pooled_tile @ W3 + b3), 8 N-tiles held in registers
      CFragF acc[8];
      #pragma unroll
      for (int nt = 0; nt < 8; ++nt)
        #pragma unroll
        for (int j = 0; j < 8; ++j) acc[nt].e[j] = 0.f;

      for (int k0 = 0; k0 < D_DIM; k0 += 4) {
        v2f a;                               // A (f32, 16x4): row=sub, K = k0+2h, k0+2h+1
        a.x = tile[wave][sub][k0 + 2 * half];
        a.y = tile[wave][sub][k0 + 2 * half + 1];
        #pragma unroll
        for (int nt = 0; nt < 8; ++nt) {
          int n = nt * 16 + sub;
          v2f bb;                            // B (f32, 4x16): col=sub, K = k0+2h, k0+2h+1
          bb.x = W3[(size_t)(k0 + 2 * half) * D_DIM + n];
          bb.y = W3[(size_t)(k0 + 2 * half + 1) * D_DIM + n];
          acc[nt].v = __builtin_amdgcn_wmma_f32_16x16x4_f32(
              false, a, false, bb, (short)0, acc[nt].v, false, false);
        }
      }

      asm volatile("s_wait_dscnt 0x0" ::: "memory");
      // bias + relu, restage h (16x128) into LDS (overwrites pooled tile)
      #pragma unroll
      for (int nt = 0; nt < 8; ++nt) {
        int n = nt * 16 + sub;
        float bias = b3[n];
        #pragma unroll
        for (int r = 0; r < 8; ++r)
          tile[wave][r + 8 * half][n] = fmaxf(acc[nt].e[r] + bias, 0.f);
      }
      asm volatile("s_wait_dscnt 0x0" ::: "memory");

      // GEMM2: logits = h @ W4 (T=12 padded to one 16-wide N tile)
      CFragF acc2;
      #pragma unroll
      for (int j = 0; j < 8; ++j) acc2.e[j] = 0.f;
      for (int k0 = 0; k0 < D_DIM; k0 += 4) {
        v2f a;
        a.x = tile[wave][sub][k0 + 2 * half];
        a.y = tile[wave][sub][k0 + 2 * half + 1];
        v2f bb;
        bb.x = (sub < T) ? W4[(size_t)(k0 + 2 * half) * T + sub] : 0.f;
        bb.y = (sub < T) ? W4[(size_t)(k0 + 2 * half + 1) * T + sub] : 0.f;
        acc2.v = __builtin_amdgcn_wmma_f32_16x16x4_f32(
            false, a, false, bb, (short)0, acc2.v, false, false);
      }
      if (sub < T) {
        float bias = b4[sub];
        #pragma unroll
        for (int r = 0; r < 8; ++r) {
          int row = rowBase + r + 8 * half;
          if (row < B) out[(size_t)row * T + sub] = acc2.e[r] + bias;
        }
      }
    }
    __syncthreads();
  }
}

// ------------------------------------------------------------------
extern "C" void kernel_launch(void* const* d_in, const int* in_sizes, int n_in,
                              void* d_out, int out_size, void* d_ws, size_t ws_size,
                              hipStream_t stream) {
  const float* feat = (const float*)d_in[0];
  const int*   fidx = (const int*)d_in[1];
  const int*   bidx = (const int*)d_in[2];
  // d_in[3] = dim_size scalar on device; B derived from out_size instead
  const float* W1 = (const float*)d_in[4];
  const float* b1 = (const float*)d_in[5];
  const float* W2 = (const float*)d_in[6];
  const float* b2 = (const float*)d_in[7];
  const float* W3 = (const float*)d_in[8];
  const float* b3 = (const float*)d_in[9];
  const float* W4 = (const float*)d_in[10];
  const float* b4 = (const float*)d_in[11];
  float* out = (float*)d_out;

  const int M = in_sizes[1];
  const int T = in_sizes[11];
  const int B = out_size / T;

  char* ws = (char*)d_ws;
  size_t off = 0;
  float* pooled = (float*)(ws + off); off += (size_t)B * D_DIM * sizeof(float);
  off = (off + 255) & ~(size_t)255;
  float* scores = (float*)(ws + off); off += (size_t)M * sizeof(float);
  off = (off + 255) & ~(size_t)255;
  _Float16* W1t = (_Float16*)(ws + off); off += (size_t)H_DIM * D_DIM * sizeof(_Float16);

  prep_w1t<<<(D_DIM * H_DIM + 255) / 256, 256, 0, stream>>>(W1, W1t);

  int numTiles = (M + 15) / 16;
  int grid1 = (numTiles + 3) / 4;
  if (grid1 > 1024) grid1 = 1024;
  attn_scores<<<grid1, 128, 0, stream>>>(feat, fidx, W1t, b1, W2, b2, scores, M);

  seg_softmax_pool<<<B, 128, 0, stream>>>(feat, fidx, bidx, scores, pooled, M);

  int rtiles = (B + 15) / 16;
  int grid3 = (rtiles + 3) / 4;
  if (grid3 > 256) grid3 = 256;
  predictor<<<grid3, 128, 0, stream>>>(pooled, W3, b3, W4, b4, out, B, T);
}
